// ProtoGCN_85014582657648
// MI455X (gfx1250) — matrix operand order repaired and checked by reading, not compiled, not run on hardware
//
#include <hip/hip_runtime.h>
#include <math.h>

typedef __bf16 bf16_t;
typedef __attribute__((ext_vector_type(8)))  __bf16 v8bf;
typedef __attribute__((ext_vector_type(16))) __bf16 v16bf;
typedef __attribute__((ext_vector_type(8)))  float  v8f;

union ABf { v16bf v; v8bf h[2]; };

// ---------------- problem constants ----------------
#define NROWS   60000
#define MP      60032          // 60000 padded to multiple of 128
#define NTRAIN  30000
#define NCLASS  64
#define KFEAT   512            // NFEAT, multiple of 32
#define CP1     2048           // H1=2000 padded to mult of 128
#define CP2     1024           // H2=1000 padded
#define CP3     384            // H3=300 padded to mult of 128
#define H1V     2000
#define H2V     1000
#define H3V     300
#define DATT    128
#define LCAP    2048           // per-class member-list capacity
#define LDSB_STRIDE 40         // bf16 units; 80B row pitch, 16B aligned

// ---------------- f32 -> bf16 convert with zero padding ----------------
__global__ void cvt_pad_kernel(const float* __restrict__ src, bf16_t* __restrict__ dst,
                               int R, int C, int Rp, int Cp, size_t total) {
    size_t stride = (size_t)gridDim.x * blockDim.x;
    for (size_t i = (size_t)blockIdx.x * blockDim.x + threadIdx.x; i < total; i += stride) {
        int r = (int)(i / Cp);
        int c = (int)(i % Cp);
        float v = (r < R && c < C) ? src[(size_t)r * C + c] : 0.0f;
        dst[i] = (bf16_t)v;
    }
}

__global__ void zero_f32_kernel(float* __restrict__ p, size_t n) {
    size_t stride = (size_t)gridDim.x * blockDim.x;
    for (size_t i = (size_t)blockIdx.x * blockDim.x + threadIdx.x; i < n; i += stride)
        p[i] = 0.0f;
}

// ---------------- bf16 WMMA GEMM, double-buffered LDS B tile ----------------
// C[M x Np] = A[M x K](bf16, lda=K) @ B[K x Np](bf16, ldb=Np)
// mode 0: store bf16 (pre-BN activations); mode 1: store f32 + bias (cols >= biasC get 0)
// Block: 256 thr = 8 waves, tile 128x128. Wave (wm=w&3, wn=w>>2): rows wm*32(+32), cols wn*64(+64).
// Per K-step per wave: 2 A frags (global), 4 B frags (LDS), 8 WMMAs.
__global__ __launch_bounds__(256) void gemm_bf16_wmma(
        const bf16_t* __restrict__ A, const bf16_t* __restrict__ B,
        void* __restrict__ Cout, int K, int Np, int mode,
        const float* __restrict__ bias, int biasC) {
    __shared__ __align__(16) bf16_t lb[2][128 * LDSB_STRIDE]; // B tiles, transposed [col][k]

    const int tid   = threadIdx.x;
    const int lane  = tid & 31;
    const int wave  = tid >> 5;
    const int wm    = wave & 3;
    const int wn    = wave >> 2;
    const int bcol0 = blockIdx.x * 128;
    const int row0  = blockIdx.y * 128 + wm * 32;
    const int l15   = lane & 15;
    const int hi    = lane >> 4;          // 0 or 1
    const int aoff  = hi << 3;            // A k-offset: 0 or 8
    const int khalf = hi << 4;            // 0 or 16; unused, B reads use aoff

    // staging decomposition: 16 threads per k-row, 8 cols each; two k-rows per thread
    const int k0s  = tid >> 4;            // 0..15
    const int cblk = (tid & 15) << 3;     // 0..120

    v8f acc[2][4];
#pragma unroll
    for (int a = 0; a < 2; ++a)
#pragma unroll
        for (int f = 0; f < 4; ++f) acc[a][f] = (v8f){};

    const bf16_t* Arow0 = A + (size_t)(row0 + l15) * K;
    const bf16_t* Arow1 = A + (size_t)(row0 + 16 + l15) * K;

    // ---- prologue: stage tile 0, load A frags for k=0 ----
    {
        v8bf t0 = *reinterpret_cast<const v8bf*>(B + (size_t)k0s * Np + bcol0 + cblk);
        v8bf t1 = *reinterpret_cast<const v8bf*>(B + (size_t)(k0s + 16) * Np + bcol0 + cblk);
#pragma unroll
        for (int i = 0; i < 8; ++i) {
            lb[0][(cblk + i) * LDSB_STRIDE + k0s]      = t0[i];
            lb[0][(cblk + i) * LDSB_STRIDE + k0s + 16] = t1[i];
        }
    }
    ABf af[2];
    af[0].h[0] = *reinterpret_cast<const v8bf*>(Arow0 + aoff);
    af[0].h[1] = *reinterpret_cast<const v8bf*>(Arow0 + 16 + aoff);
    af[1].h[0] = *reinterpret_cast<const v8bf*>(Arow1 + aoff);
    af[1].h[1] = *reinterpret_cast<const v8bf*>(Arow1 + 16 + aoff);
    __syncthreads();

    const int nk = K >> 5;
    for (int it = 0; it < nk; ++it) {
        const int  cb      = it & 1;
        const int  nb      = cb ^ 1;
        const int  kn      = (it + 1) << 5;
        const bool hasnext = (it + 1) < nk;

        // issue next B tile global loads early (latency hidden under WMMAs)
        v8bf t0 = {}, t1 = {};
        if (hasnext) {
            t0 = *reinterpret_cast<const v8bf*>(B + (size_t)(kn + k0s) * Np + bcol0 + cblk);
            t1 = *reinterpret_cast<const v8bf*>(B + (size_t)(kn + k0s + 16) * Np + bcol0 + cblk);
        }
        // prefetch next A fragments into registers
        ABf afn[2];
        if (hasnext) {
            afn[0].h[0] = *reinterpret_cast<const v8bf*>(Arow0 + kn + aoff);
            afn[0].h[1] = *reinterpret_cast<const v8bf*>(Arow0 + kn + 16 + aoff);
            afn[1].h[0] = *reinterpret_cast<const v8bf*>(Arow1 + kn + aoff);
            afn[1].h[1] = *reinterpret_cast<const v8bf*>(Arow1 + kn + 16 + aoff);
        }
        // all four B fragments in distinct registers so ds_loads cluster ahead of WMMAs
        ABf bfr[4];
#pragma unroll
        for (int f = 0; f < 4; ++f) {
            const bf16_t* lp = &lb[cb][(wn * 64 + f * 16 + l15) * LDSB_STRIDE + aoff];
            bfr[f].h[0] = *reinterpret_cast<const v8bf*>(lp);
            bfr[f].h[1] = *reinterpret_cast<const v8bf*>(lp + 16);
        }
#pragma unroll
        for (int a = 0; a < 2; ++a)
#pragma unroll
            for (int f = 0; f < 4; ++f)
                acc[a][f] = __builtin_amdgcn_wmma_f32_16x16x32_bf16(
                    false, af[a].v, false, bfr[f].v, (short)0, acc[a][f], false, false);

        if (hasnext) {
#pragma unroll
            for (int i = 0; i < 8; ++i) {
                lb[nb][(cblk + i) * LDSB_STRIDE + k0s]      = t0[i];
                lb[nb][(cblk + i) * LDSB_STRIDE + k0s + 16] = t1[i];
            }
        }
        __syncthreads();
        if (hasnext) { af[0] = afn[0]; af[1] = afn[1]; }
    }

    // ---- epilogue: lane -> col bcol0+wn*64+f*16+l15, rows row0+a*16+hi*8+v ----
    if (mode == 0) {
        bf16_t* C = (bf16_t*)Cout;
#pragma unroll
        for (int a = 0; a < 2; ++a) {
            int rbase = row0 + a * 16 + (hi << 3);
#pragma unroll
            for (int f = 0; f < 4; ++f) {
                int col = bcol0 + wn * 64 + f * 16 + l15;
#pragma unroll
                for (int v = 0; v < 8; ++v)
                    C[(size_t)(rbase + v) * Np + col] = (bf16_t)acc[a][f][v];
            }
        }
    } else {
        float* C = (float*)Cout;
#pragma unroll
        for (int a = 0; a < 2; ++a) {
            int rbase = row0 + a * 16 + (hi << 3);
#pragma unroll
            for (int f = 0; f < 4; ++f) {
                int col = bcol0 + wn * 64 + f * 16 + l15;
                float bb = (col < biasC) ? bias[col] : 0.0f;
#pragma unroll
                for (int v = 0; v < 8; ++v)
                    C[(size_t)(rbase + v) * Np + col] = acc[a][f][v] + bb;
            }
        }
    }
}

// ---------------- column stats (sum, sumsq) over first M rows ----------------
__global__ __launch_bounds__(256) void colstats_kernel(
        const bf16_t* __restrict__ Z, float* __restrict__ colsum,
        float* __restrict__ colsumsq, int M, int Np) {
    int c = blockIdx.x * blockDim.x + threadIdx.x;
    if (c >= Np) return;
    int r0 = blockIdx.y * 1024;
    int r1 = r0 + 1024; if (r1 > M) r1 = M;
    float s = 0.0f, s2 = 0.0f;
    for (int r = r0; r < r1; ++r) {
        float v = (float)Z[(size_t)r * Np + c];
        s += v; s2 = fmaf(v, v, s2);
    }
    atomicAdd(&colsum[c], s);
    atomicAdd(&colsumsq[c], s2);
}

// ---------------- fold BN into per-column scale/shift ----------------
__global__ void bn_coef_kernel(const float* __restrict__ colsum,
                               const float* __restrict__ colsumsq,
                               const float* __restrict__ g,
                               const float* __restrict__ be,
                               float* __restrict__ ca, float* __restrict__ cb,
                               int C, int Np, float invM) {
    int c = blockIdx.x * blockDim.x + threadIdx.x;
    if (c >= Np) return;
    if (c < C) {
        float m   = colsum[c] * invM;
        float var = fmaf(-m, m, colsumsq[c] * invM);
        float a   = g[c] * rsqrtf(var + 1e-5f);
        ca[c] = a;
        cb[c] = fmaf(-m, a, be[c]);
    } else {
        ca[c] = 0.0f;   // pad columns stay exactly zero
        cb[c] = 0.0f;
    }
}

// ---------------- in-place y = leaky(z*a[c] + b[c]) (bf16) ----------------
__global__ void bn_apply_kernel(bf16_t* __restrict__ Z,
                                const float* __restrict__ ca,
                                const float* __restrict__ cb,
                                int Np, size_t total) {
    size_t stride = (size_t)gridDim.x * blockDim.x;
    for (size_t i = (size_t)blockIdx.x * blockDim.x + threadIdx.x; i < total; i += stride) {
        int c   = (int)(i % Np);
        float y = fmaf((float)Z[i], ca[c], cb[c]);
        y = (y >= 0.0f) ? y : 0.01f * y;
        Z[i] = (bf16_t)y;
    }
}

// ---------------- attention scores s[i] = tanh(h3[row]@Wa + ba)@wv + bv ----------------
__global__ __launch_bounds__(128) void attn_kernel(
        const float* __restrict__ h3, const int* __restrict__ idx_train,
        const float* __restrict__ Wa, const float* __restrict__ ba,
        const float* __restrict__ wv, const float* __restrict__ bv,
        float* __restrict__ s_out) {
    __shared__ float hrow[H3V];
    __shared__ float red[DATT];
    int i = blockIdx.x, t = threadIdx.x;
    int row = idx_train[i];
    for (int k = t; k < H3V; k += DATT) hrow[k] = h3[(size_t)row * CP3 + k];
    __syncthreads();
    float acc = 0.0f;
    for (int k = 0; k < H3V; ++k) acc = fmaf(hrow[k], Wa[k * DATT + t], acc);
    red[t] = tanhf(acc + ba[t]) * wv[t];
    __syncthreads();
    for (int st = DATT / 2; st > 0; st >>= 1) {
        if (t < st) red[t] += red[t + st];
        __syncthreads();
    }
    if (t == 0) s_out[i] = red[0] + bv[0];
}

// ---------------- per-class member lists ----------------
__global__ void build_lists_kernel(const int* __restrict__ labels,
                                   const int* __restrict__ idx_train,
                                   int* __restrict__ counts, int* __restrict__ lists) {
    int i = blockIdx.x * blockDim.x + threadIdx.x;
    if (i >= NTRAIN) return;
    int lbl = labels[idx_train[i]];
    if (lbl < 0 || lbl >= NCLASS) return;
    int pos = atomicAdd(&counts[lbl], 1);
    if (pos < LCAP) lists[lbl * LCAP + pos] = i;
}

// ---------------- per-class softmax stats: max and sum of exp ----------------
__global__ __launch_bounds__(256) void class_stats_kernel(
        const float* __restrict__ s, const int* __restrict__ counts,
        const int* __restrict__ lists, float* __restrict__ mx, float* __restrict__ sumexp) {
    __shared__ float red[256];
    int c = blockIdx.x, t = threadIdx.x;
    int cnt = counts[c]; if (cnt > LCAP) cnt = LCAP;
    float lm = -3.4e38f;
    for (int j = t; j < cnt; j += 256) lm = fmaxf(lm, s[lists[c * LCAP + j]]);
    red[t] = lm; __syncthreads();
    for (int st = 128; st > 0; st >>= 1) {
        if (t < st) red[t] = fmaxf(red[t], red[t + st]);
        __syncthreads();
    }
    float m = red[0];
    __syncthreads();
    float ls = 0.0f;
    for (int j = t; j < cnt; j += 256) ls += expf(s[lists[c * LCAP + j]] - m);
    red[t] = ls; __syncthreads();
    for (int st = 128; st > 0; st >>= 1) {
        if (t < st) red[t] += red[t + st];
        __syncthreads();
    }
    if (t == 0) {
        mx[c]     = (cnt > 0) ? m      : 0.0f;
        sumexp[c] = (cnt > 0) ? red[0] : 1.0f;
    }
}

// ---------------- prototype accumulation: proto[c] = sum_i w_i * h3[row_i] ----------------
__global__ __launch_bounds__(CP3) void proto_kernel(
        const float* __restrict__ h3, const float* __restrict__ s,
        const int* __restrict__ idx_train, const int* __restrict__ counts,
        const int* __restrict__ lists, const float* __restrict__ mx,
        const float* __restrict__ sumexp, float* __restrict__ proto) {
    int c = blockIdx.x, d = threadIdx.x;
    int cnt = counts[c]; if (cnt > LCAP) cnt = LCAP;
    float m = mx[c];
    float inv = 1.0f / sumexp[c];
    float acc = 0.0f;
    if (d < H3V) {
        for (int j = 0; j < cnt; ++j) {
            int i = lists[c * LCAP + j];
            float w = expf(s[i] - m) * inv;
            acc = fmaf(w, h3[(size_t)idx_train[i] * CP3 + d], acc);
        }
    }
    proto[c * CP3 + d] = acc;
}

// ---------------- distances: out[n,c] = -sqrt(max(||h_n - p_c||^2, 1e-12)) ----------------
__global__ __launch_bounds__(256) void dist_kernel(
        const float* __restrict__ h3, const float* __restrict__ proto,
        float* __restrict__ out) {
    int n = blockIdx.x * 4 + threadIdx.y;
    int c = threadIdx.x;
    const float* hp = h3 + (size_t)n * CP3;
    const float* pp = proto + c * CP3;
    float sq = 0.0f;
#pragma unroll 5
    for (int k = 0; k < H3V; ++k) {
        float d = hp[k] - pp[k];
        sq = fmaf(d, d, sq);
    }
    out[(size_t)n * NCLASS + c] = -sqrtf(fmaxf(sq, 1e-12f));
}

// ---------------- host-side launch ----------------
extern "C" void kernel_launch(void* const* d_in, const int* in_sizes, int n_in,
                              void* d_out, int out_size, void* d_ws, size_t ws_size,
                              hipStream_t stream) {
    const float* x      = (const float*)d_in[0];
    const int*   labels = (const int*)  d_in[1];
    const int*   idx_tr = (const int*)  d_in[2];
    const float* W1     = (const float*)d_in[5];
    const float* g1     = (const float*)d_in[7];
    const float* be1    = (const float*)d_in[8];
    const float* W2     = (const float*)d_in[9];
    const float* g2     = (const float*)d_in[11];
    const float* be2    = (const float*)d_in[12];
    const float* W3     = (const float*)d_in[13];
    const float* b3     = (const float*)d_in[14];
    const float* Wa     = (const float*)d_in[15];
    const float* ba     = (const float*)d_in[16];
    const float* wv     = (const float*)d_in[17];
    const float* bv     = (const float*)d_in[18];
    float* out = (float*)d_out;

    // workspace carve-out (256B aligned sections)
    size_t off = 0;
    auto carve = [&](size_t bytes) -> void* {
        void* p = (char*)d_ws + off;
        off = (off + bytes + 255) & ~(size_t)255;
        return p;
    };
    bf16_t* Xb    = (bf16_t*)carve((size_t)MP * KFEAT * 2);
    bf16_t* W1b   = (bf16_t*)carve((size_t)KFEAT * CP1 * 2);
    bf16_t* W2b   = (bf16_t*)carve((size_t)CP1 * CP2 * 2);
    bf16_t* W3b   = (bf16_t*)carve((size_t)CP2 * CP3 * 2);
    bf16_t* ZH1   = (bf16_t*)carve((size_t)MP * CP1 * 2);
    bf16_t* ZH2   = (bf16_t*)carve((size_t)MP * CP2 * 2);
    float*  H3    = (float*) carve((size_t)MP * CP3 * 4);
    float*  sbuf  = (float*) carve((size_t)NTRAIN * 4);
    float*  csum  = (float*) carve((size_t)CP1 * 4);
    float*  csum2 = (float*) carve((size_t)CP1 * 4);
    float*  ca    = (float*) carve((size_t)CP1 * 4);
    float*  cbv   = (float*) carve((size_t)CP1 * 4);
    int*    cnts  = (int*)   carve((size_t)NCLASS * 4);
    int*    lists = (int*)   carve((size_t)NCLASS * LCAP * 4);
    float*  mx    = (float*) carve((size_t)NCLASS * 4);
    float*  sume  = (float*) carve((size_t)NCLASS * 4);
    float*  proto = (float*) carve((size_t)NCLASS * CP3 * 4);
    (void)ws_size; (void)in_sizes; (void)n_in; (void)out_size;

    const int CB = 256;
    const int GS = 8192; // grid-stride launch width

    // 1) bf16 conversions with padding
    cvt_pad_kernel<<<GS, CB, 0, stream>>>(x,  Xb,  NROWS, KFEAT, MP,    KFEAT, (size_t)MP * KFEAT);
    cvt_pad_kernel<<<GS, CB, 0, stream>>>(W1, W1b, KFEAT, H1V,   KFEAT, CP1,   (size_t)KFEAT * CP1);
    cvt_pad_kernel<<<GS, CB, 0, stream>>>(W2, W2b, H1V,   H2V,   CP1,   CP2,   (size_t)CP1 * CP2);
    cvt_pad_kernel<<<GS, CB, 0, stream>>>(W3, W3b, H2V,   H3V,   CP2,   CP3,   (size_t)CP2 * CP3);

    // 2) Layer 1: Z1 = Xb @ W1b  (bias cancels inside BN)
    zero_f32_kernel<<<16, CB, 0, stream>>>(csum,  CP1);
    zero_f32_kernel<<<16, CB, 0, stream>>>(csum2, CP1);
    gemm_bf16_wmma<<<dim3(CP1 / 128, MP / 128), CB, 0, stream>>>(
        Xb, W1b, ZH1, KFEAT, CP1, 0, nullptr, 0);
    colstats_kernel<<<dim3((CP1 + CB - 1) / CB, (NROWS + 1023) / 1024), CB, 0, stream>>>(
        ZH1, csum, csum2, NROWS, CP1);
    bn_coef_kernel<<<(CP1 + CB - 1) / CB, CB, 0, stream>>>(
        csum, csum2, g1, be1, ca, cbv, H1V, CP1, 1.0f / NROWS);
    bn_apply_kernel<<<GS, CB, 0, stream>>>(ZH1, ca, cbv, CP1, (size_t)MP * CP1);

    // 3) Layer 2: Z2 = H1 @ W2b
    zero_f32_kernel<<<16, CB, 0, stream>>>(csum,  CP2);
    zero_f32_kernel<<<16, CB, 0, stream>>>(csum2, CP2);
    gemm_bf16_wmma<<<dim3(CP2 / 128, MP / 128), CB, 0, stream>>>(
        ZH1, W2b, ZH2, CP1, CP2, 0, nullptr, 0);
    colstats_kernel<<<dim3((CP2 + CB - 1) / CB, (NROWS + 1023) / 1024), CB, 0, stream>>>(
        ZH2, csum, csum2, NROWS, CP2);
    bn_coef_kernel<<<(CP2 + CB - 1) / CB, CB, 0, stream>>>(
        csum, csum2, g2, be2, ca, cbv, H2V, CP2, 1.0f / NROWS);
    bn_apply_kernel<<<GS, CB, 0, stream>>>(ZH2, ca, cbv, CP2, (size_t)MP * CP2);

    // 4) Layer 3: H3 = H2 @ W3b + b3  (f32 output)
    gemm_bf16_wmma<<<dim3(CP3 / 128, MP / 128), CB, 0, stream>>>(
        ZH2, W3b, H3, CP2, CP3, 1, b3, H3V);

    // 5) attention scores on training rows
    attn_kernel<<<NTRAIN, DATT, 0, stream>>>(H3, idx_tr, Wa, ba, wv, bv, sbuf);

    // 6) per-class softmax -> prototypes
    zero_f32_kernel<<<1, 64, 0, stream>>>((float*)cnts, NCLASS);
    build_lists_kernel<<<(NTRAIN + CB - 1) / CB, CB, 0, stream>>>(labels, idx_tr, cnts, lists);
    class_stats_kernel<<<NCLASS, 256, 0, stream>>>(sbuf, cnts, lists, mx, sume);
    proto_kernel<<<NCLASS, CP3, 0, stream>>>(H3, sbuf, idx_tr, cnts, lists, mx, sume, proto);

    // 7) distances -> output
    dist_kernel<<<NROWS / 4, dim3(NCLASS, 4), 0, stream>>>(H3, proto, out);
}
